// _RNN_223338300164
// MI455X (gfx1250) — matrix operand order repaired and checked by reading, not compiled, not run on hardware
//
#include <hip/hip_runtime.h>
#include <stdint.h>

// Problem constants (match reference)
#define HH     512      // state_hsize
#define NGATE  2048     // 4*H, gate-interleaved: column 4j+g
#define TT     256
#define BBATCH 128
#define D_IN   128
#define OUTW   48

#define LDSW   20       // LDS A-tile row stride in dwords (80B: 16B-aligned rows,
                        // bank-conflict-free fragment reads)

typedef __attribute__((ext_vector_type(16))) __bf16 v16bf;
typedef __attribute__((ext_vector_type(8)))  float  v8f;

union Frag16 { v16bf v; unsigned int u[8]; };

__device__ __forceinline__ unsigned short f2bf(float f) {
  unsigned int u = __float_as_uint(f);
  unsigned int r = 0x7FFFu + ((u >> 16) & 1u);   // round-to-nearest-even
  return (unsigned short)((u + r) >> 16);
}
__device__ __forceinline__ unsigned int pack2bf(float a, float b) {
  return (unsigned int)f2bf(a) | ((unsigned int)f2bf(b) << 16);
}
__device__ __forceinline__ float sigf(float x) { return 1.0f / (1.0f + __expf(-x)); }

// CDNA5 async LDS fill: 16B from global -> LDS, tracked by ASYNCcnt.
// VDST = 32-bit LDS byte address (low 32 bits of the flat shared pointer),
// VADDR = 64-bit global address (ISA 15.18.3, GLOBAL_LOAD_ASYNC_TO_LDS_B128).
__device__ __forceinline__ void async_lds_b128(unsigned int* ldsDst, const void* gsrc) {
  unsigned lds = (unsigned)(size_t)(void*)ldsDst;
  asm volatile("global_load_async_to_lds_b128 %0, %1, off"
               :: "v"(lds), "v"(gsrc) : "memory");
}
__device__ __forceinline__ void wait_async0() {
  asm volatile("s_wait_asynccnt 0x0" ::: "memory");
}

// ---------------------------------------------------------------------------
// Weight prep: gate-interleave + concat [W_ih | W_hh] -> bf16 (NGATE x K),
// N-major (row n, stride K) so B-fragment K-pairs are contiguous dwords.
// ---------------------------------------------------------------------------
__global__ __launch_bounds__(256) void prep_lstm_w(
    const float* __restrict__ Wih, const float* __restrict__ Whh,
    const float* __restrict__ bih, const float* __restrict__ bhh,
    int Din, int K, unsigned short* __restrict__ Wc, float* __restrict__ bc)
{
  int idx = blockIdx.x * 256 + threadIdx.x;
  if (idx >= NGATE * K) return;
  int n = idx / K, k = idx - n * K;
  int j = n >> 2, g = n & 3;           // hidden unit, gate (i,f,g,o)
  int r0 = g * HH + j;                 // original row in (4H x *) weight
  float val = (k < Din) ? Wih[(size_t)r0 * Din + k]
                        : Whh[(size_t)r0 * HH + (k - Din)];
  Wc[idx] = f2bf(val);
  if (k == 0) bc[n] = bih[r0] + bhh[r0];
}

// Generic f32 -> bf16 convert with zero row padding (N-major, row stride K).
__global__ __launch_bounds__(256) void prep_pad_w(
    const float* __restrict__ src, int Nsrc, int K, int Npad,
    unsigned short* __restrict__ dst)
{
  int idx = blockIdx.x * 256 + threadIdx.x;
  if (idx >= Npad * K) return;
  int n = idx / K, k = idx - n * K;
  dst[idx] = (n < Nsrc) ? f2bf(src[(size_t)n * K + k]) : (unsigned short)0;
}

// Flat f32 -> bf16 cast (for the network input).
__global__ __launch_bounds__(256) void prep_cast(
    const float* __restrict__ src, int count, unsigned short* __restrict__ dst)
{
  int idx = blockIdx.x * 256 + threadIdx.x;
  if (idx < count) dst[idx] = f2bf(src[idx]);
}

// ---------------------------------------------------------------------------
// One fused LSTM step:  g = [x_t | h_{t-1}] @ Wc^T + b ; gate math ; write h,c.
// All activations are bf16, so A-tile staging is a pure 16B copy done with
// async LDS loads. Block: 128(M) x 64(N); 8 waves, each 16(M) x 64(N).
// Grid: (NGATE/64, B2/128). hprev == nullptr encodes t == 0 (h=c=0) and the
// host passes Kloop = Din so the zero h columns are skipped entirely.
// Din is a multiple of 32, so each 32-wide K-step is wholly x or wholly h.
// ---------------------------------------------------------------------------
__global__ __launch_bounds__(256) void lstm_step_kernel(
    const unsigned short* __restrict__ xslab, int Din,
    const unsigned short* __restrict__ hprev,
    const unsigned short* __restrict__ Wc, int Kstride, int Kloop,
    const float* __restrict__ bc,
    float* __restrict__ cbuf,
    unsigned short* __restrict__ hout)
{
  __shared__ unsigned int Asu[128 * LDSW];        // 128 x 32 bf16 tile
  const int tid  = threadIdx.x;
  const int lane = tid & 31;
  const int wave = tid >> 5;
  const int bm = blockIdx.y, bn = blockIdx.x;

  v8f acc[4];
  #pragma unroll
  for (int t = 0; t < 4; ++t)
    #pragma unroll
    for (int i = 0; i < 8; ++i) acc[t][i] = 0.0f;

  for (int kk = 0; kk < Kloop; kk += 32) {
    // Uniform (scalar) source select for this whole K-step.
    const unsigned short* src;
    int stride, kbase;
    if (kk < Din) { src = xslab; stride = Din; kbase = kk; }
    else          { src = hprev; stride = HH;  kbase = kk - Din; }

    __syncthreads();                    // WAR: consumers of previous tile done
    // Async-stage A tile: 512 b128 chunks (row, 4 chunks of 8 bf16).
    #pragma unroll
    for (int s = 0; s < 2; ++s) {
      int ci = tid + s * 256;
      int row = ci >> 2, c = ci & 3;
      const unsigned short* g =
          src + (size_t)(bm * 128 + row) * stride + kbase + c * 8;
      async_lds_b128(&Asu[row * LDSW + c * 4], g);
    }
    wait_async0();
    __syncthreads();

    // A fragment (16x32 bf16): lane m = lane%16; reg i -> K pair 16*(lane/16)+2i.
    Frag16 a;
    int am = (wave << 4) + (lane & 15);
    #pragma unroll
    for (int i = 0; i < 8; ++i) {
      int kd = ((lane & 16) ? 8 : 0) + i; // dword in row
      a.u[i] = Asu[am * LDSW + kd];
    }
    // 4 N-tiles: B fragment (32x16 bf16) straight from L2-resident weights.
    #pragma unroll
    for (int t = 0; t < 4; ++t) {
      Frag16 b;
      int n = bn * 64 + t * 16 + (lane & 15);
      const unsigned short* wbase =
          Wc + (size_t)n * Kstride + kk + ((lane & 16) ? 16 : 0);
      const unsigned int* wp = (const unsigned int*)wbase;
      #pragma unroll
      for (int i = 0; i < 8; ++i) b.u[i] = wp[i];
      // Locality 3 -> WGP-scope prefetch: fills L0/WGP$ (SYS scope would stop
      // at GL2, where the weights already live).
      __builtin_prefetch((const char*)wbase + 64, 0, 3);
      acc[t] = __builtin_amdgcn_wmma_f32_16x16x32_bf16(
          false, a.v, false, b.v, (short)0, acc[t], false, false);
    }
  }

  // Epilogue: bias, gather i/f/g/o across the lane quad, LSTM cell update.
  const int lm = lane & 15;
  #pragma unroll
  for (int t = 0; t < 4; ++t) {
    int n_g = bn * 64 + t * 16 + lm;      // gate-interleaved column = 4j+g
    float bias = bc[n_g];
    #pragma unroll
    for (int i = 0; i < 8; ++i) {
      float gv = acc[t][i] + bias;
      float vi = __shfl(gv, (lane & ~3) | 0, 32);
      float vf = __shfl(gv, (lane & ~3) | 1, 32);
      float vg = __shfl(gv, (lane & ~3) | 2, 32);
      float vo = __shfl(gv, (lane & ~3) | 3, 32);
      if ((lane & 3) == 0) {
        int j  = n_g >> 2;                                   // hidden unit
        int gr = bm * 128 + (wave << 4) + i + ((lane & 16) ? 8 : 0);
        float cprev = hprev ? cbuf[(size_t)gr * HH + j] : 0.0f;
        float cg = sigf(vf) * cprev + sigf(vi) * tanhf(vg);
        float hv = sigf(vo) * tanhf(cg);
        cbuf[(size_t)gr * HH + j] = cg;
        hout[(size_t)gr * HH + j] = f2bf(hv);
      }
    }
  }
}

// ---------------------------------------------------------------------------
// MLP: y = tanh((h3 + h1) @ mlp_w^T + b) -> bf16 (feeds adapter directly).
// Fuses the stack-1 residual on bf16 inputs. M=32768, N=K=512. Grid (8, 256).
// ---------------------------------------------------------------------------
__global__ __launch_bounds__(256) void mlp_kernel(
    const unsigned short* __restrict__ xa, const unsigned short* __restrict__ xb,
    const unsigned short* __restrict__ Wb, const float* __restrict__ bias,
    unsigned short* __restrict__ ybf)
{
  __shared__ unsigned int Asu[128 * LDSW];
  const int tid = threadIdx.x, lane = tid & 31, wave = tid >> 5;
  const int bm = blockIdx.y, bn = blockIdx.x;
  const int K = HH;

  v8f acc[4];
  #pragma unroll
  for (int t = 0; t < 4; ++t)
    #pragma unroll
    for (int i = 0; i < 8; ++i) acc[t][i] = 0.0f;

  for (int kk = 0; kk < K; kk += 32) {
    __syncthreads();
    #pragma unroll
    for (int idx = tid; idx < 128 * 16; idx += 256) {
      int row = idx >> 4, cp = idx & 15;
      size_t base = (size_t)(bm * 128 + row) * HH + kk + cp * 2;
      unsigned int A = *(const unsigned int*)(xa + base);
      unsigned int B = *(const unsigned int*)(xb + base);
      float a0 = __uint_as_float(A << 16), a1 = __uint_as_float(A & 0xFFFF0000u);
      float b0 = __uint_as_float(B << 16), b1 = __uint_as_float(B & 0xFFFF0000u);
      Asu[row * LDSW + cp] = pack2bf(a0 + b0, a1 + b1);
    }
    __syncthreads();
    Frag16 a;
    int am = (wave << 4) + (lane & 15);
    #pragma unroll
    for (int i = 0; i < 8; ++i) {
      int kd = ((lane & 16) ? 8 : 0) + i;
      a.u[i] = Asu[am * LDSW + kd];
    }
    #pragma unroll
    for (int t = 0; t < 4; ++t) {
      Frag16 b;
      int n = bn * 64 + t * 16 + (lane & 15);
      const unsigned short* wbase =
          Wb + (size_t)n * K + kk + ((lane & 16) ? 16 : 0);
      const unsigned int* wp = (const unsigned int*)wbase;
      #pragma unroll
      for (int i = 0; i < 8; ++i) b.u[i] = wp[i];
      __builtin_prefetch((const char*)wbase + 64, 0, 3);
      acc[t] = __builtin_amdgcn_wmma_f32_16x16x32_bf16(
          false, a.v, false, b.v, (short)0, acc[t], false, false);
    }
  }
  const int lm = lane & 15;
  #pragma unroll
  for (int t = 0; t < 4; ++t) {
    int n_g = bn * 64 + t * 16 + lm;
    float bsv = bias[n_g];
    #pragma unroll
    for (int i = 0; i < 8; ++i) {
      int gr = bm * 128 + (wave << 4) + i + ((lane & 16) ? 8 : 0);
      ybf[(size_t)gr * HH + n_g] = f2bf(tanhf(acc[t][i] + bsv));
    }
  }
}

// ---------------------------------------------------------------------------
// Adapter: out = y @ adapter_w^T + b. N=48 padded to one 64-wide tile.
// A already bf16 -> async-staged raw. Grid (1, 256).
// ---------------------------------------------------------------------------
__global__ __launch_bounds__(256) void adapter_kernel(
    const unsigned short* __restrict__ ybf,
    const unsigned short* __restrict__ Wb,     // 64 x 512 (rows 48..63 zero)
    const float* __restrict__ bias, float* __restrict__ out)
{
  __shared__ unsigned int Asu[128 * LDSW];
  const int tid = threadIdx.x, lane = tid & 31, wave = tid >> 5;
  const int bm = blockIdx.y;
  const int K = HH;

  v8f acc[4];
  #pragma unroll
  for (int t = 0; t < 4; ++t)
    #pragma unroll
    for (int i = 0; i < 8; ++i) acc[t][i] = 0.0f;

  for (int kk = 0; kk < K; kk += 32) {
    __syncthreads();
    #pragma unroll
    for (int s = 0; s < 2; ++s) {
      int ci = tid + s * 256;
      int row = ci >> 2, c = ci & 3;
      const unsigned short* g =
          ybf + (size_t)(bm * 128 + row) * HH + kk + c * 8;
      async_lds_b128(&Asu[row * LDSW + c * 4], g);
    }
    wait_async0();
    __syncthreads();
    Frag16 a;
    int am = (wave << 4) + (lane & 15);
    #pragma unroll
    for (int i = 0; i < 8; ++i) {
      int kd = ((lane & 16) ? 8 : 0) + i;
      a.u[i] = Asu[am * LDSW + kd];
    }
    #pragma unroll
    for (int t = 0; t < 4; ++t) {
      Frag16 b;
      int n = t * 16 + (lane & 15);
      const unsigned int* wp = (const unsigned int*)
          (Wb + (size_t)n * K + kk + ((lane & 16) ? 16 : 0));
      #pragma unroll
      for (int i = 0; i < 8; ++i) b.u[i] = wp[i];
      acc[t] = __builtin_amdgcn_wmma_f32_16x16x32_bf16(
          false, a.v, false, b.v, (short)0, acc[t], false, false);
    }
  }
  const int lm = lane & 15;
  #pragma unroll
  for (int t = 0; t < 4; ++t) {
    int n_g = t * 16 + lm;
    if (n_g < OUTW) {
      float bsv = bias[n_g];
      #pragma unroll
      for (int i = 0; i < 8; ++i) {
        int gr = bm * 128 + (wave << 4) + i + ((lane & 16) ? 8 : 0);
        out[(size_t)gr * OUTW + n_g] = acc[t][i] + bsv;
      }
    }
  }
}

// ---------------------------------------------------------------------------
extern "C" void kernel_launch(void* const* d_in, const int* in_sizes, int n_in,
                              void* d_out, int out_size, void* d_ws, size_t ws_size,
                              hipStream_t stream)
{
  (void)in_sizes; (void)n_in; (void)out_size; (void)ws_size;
  const float* input = (const float*)d_in[0];
  const float *Wih[4], *Whh[4], *bih[4], *bhh[4];
  for (int i = 0; i < 4; ++i) {
    Wih[i] = (const float*)d_in[1 + 4 * i];
    Whh[i] = (const float*)d_in[2 + 4 * i];
    bih[i] = (const float*)d_in[3 + 4 * i];
    bhh[i] = (const float*)d_in[4 + 4 * i];
  }
  const float* mlp_w = (const float*)d_in[17];
  const float* mlp_b = (const float*)d_in[18];
  const float* ad_w  = (const float*)d_in[19];
  const float* ad_b  = (const float*)d_in[20];
  float* out = (float*)d_out;

  // Workspace carve-out (~155 MB total; bf16 activations).
  char*  ws  = (char*)d_ws;
  size_t off = 0;
  auto carve = [&](size_t bytes) -> void* {
    void* p = ws + off;
    off = (off + bytes + 255) & ~(size_t)255;
    return p;
  };
  int Kc[4] = { D_IN + HH, 2 * HH, 2 * HH, 2 * HH };   // 640, 1024 x3
  unsigned short* Wc[4];
  for (int i = 0; i < 4; ++i) Wc[i] = (unsigned short*)carve((size_t)NGATE * Kc[i] * 2);
  float* bc[4];
  for (int i = 0; i < 4; ++i) bc[i] = (float*)carve((size_t)NGATE * 4);
  unsigned short* mlpWb = (unsigned short*)carve((size_t)HH * HH * 2);
  unsigned short* adWb  = (unsigned short*)carve((size_t)64 * HH * 2);
  float* cbuf = (float*)carve((size_t)1024 * HH * 4);
  const size_t seqElems = (size_t)TT * BBATCH * HH;
  unsigned short* SA  = (unsigned short*)carve(seqElems * 2);
  unsigned short* SB  = (unsigned short*)carve(seqElems * 2);
  unsigned short* SC  = (unsigned short*)carve(seqElems * 2);
  unsigned short* ybf = (unsigned short*)carve(seqElems * 2);
  unsigned short* x0b = (unsigned short*)carve((size_t)TT * BBATCH * D_IN * 2);

  // Weight / input prep (re-done every call; deterministic).
  for (int i = 0; i < 4; ++i) {
    int total = NGATE * Kc[i];
    prep_lstm_w<<<(total + 255) / 256, 256, 0, stream>>>(
        Wih[i], Whh[i], bih[i], bhh[i], Kc[i] - HH, Kc[i], Wc[i], bc[i]);
  }
  prep_pad_w<<<(HH * HH + 255) / 256, 256, 0, stream>>>(mlp_w, HH, HH, HH, mlpWb);
  prep_pad_w<<<(64 * HH + 255) / 256, 256, 0, stream>>>(ad_w, OUTW, HH, 64, adWb);
  {
    int cnt = TT * BBATCH * D_IN;
    prep_cast<<<(cnt + 255) / 256, 256, 0, stream>>>(input, cnt, x0b);
  }

  // Dilated LSTM layers. Contiguous-slab trick: step t of a rate-r layer
  // reads rows [t*r*B, (t+1)*r*B) of the input sequence and of its own
  // previous output (h_{t-1}); writes the same rows of its output.
  struct Layer { const unsigned short* in; int Din; unsigned short* outSeq; int r; int cell; };
  Layer layers[4] = {
    { x0b, D_IN, SA, 1, 0 },
    { SA,  HH,   SB, 2, 1 },
    { SB,  HH,   SC, 4, 2 },
    { SC,  HH,   SA, 8, 3 },     // SA (h_seq0) dead after layer 1
  };
  for (int L = 0; L < 4; ++L) {
    const Layer& ly = layers[L];
    int B2 = ly.r * BBATCH;
    int steps = TT / ly.r;                  // 256 divides all rates: no padding
    dim3 grid(NGATE / 64, B2 / 128);
    for (int t = 0; t < steps; ++t) {
      const unsigned short* xs = ly.in + (size_t)t * B2 * ly.Din;
      const unsigned short* hp =
          (t == 0) ? nullptr : ly.outSeq + (size_t)(t - 1) * B2 * HH;
      unsigned short* ho = ly.outSeq + (size_t)t * B2 * HH;
      int Kloop = (t == 0) ? ly.Din : Kc[ly.cell];   // skip zero h cols at t=0
      lstm_step_kernel<<<grid, 256, 0, stream>>>(
          xs, ly.Din, hp, Wc[ly.cell], Kc[ly.cell], Kloop, bc[ly.cell], cbuf, ho);
    }
  }

  // MLP with fused residual (h3 + h1), then adapter to d_out.
  mlp_kernel<<<dim3(HH / 64, (TT * BBATCH) / 128), 256, 0, stream>>>(
      SA, SB, mlpWb, mlp_b, ybf);
  adapter_kernel<<<dim3(1, (TT * BBATCH) / 128), 256, 0, stream>>>(
      ybf, adWb, ad_b, out);
}
